// ContrastiveLoss_46471546142761
// MI455X (gfx1250) — compile-verified, matrix operand order checked by reading, not run on hardware
//
#include <hip/hip_runtime.h>
#include <stdint.h>

// ---------------------------------------------------------------------------
// Problem constants (reference: 8192 x 512 fp32 embeddings, T = 0.05)
// ---------------------------------------------------------------------------
#define N_ROWS 8192
#define N_DIM  512
#define INV_T  20.0f
#define EPSV   1e-8f

typedef __bf16 bf16_t;
typedef bf16_t v16bf __attribute__((ext_vector_type(16)));
typedef float  v8f   __attribute__((ext_vector_type(8)));

union FragU {
    uint4 u4[2];   // two 16-byte chunks = 16 bf16 elements
    v16bf v;
};

__device__ __forceinline__ void load_frag(FragU& f, const unsigned short* p) {
    f.u4[0] = *(const uint4*)p;
    f.u4[1] = *(const uint4*)(p + 16);
}

__device__ __forceinline__ unsigned int pack_bf16_rne(float lo, float hi) {
    union { float f; uint32_t u; } a, b;
    a.f = lo; b.f = hi;
    uint32_t ra = a.u + 0x7FFFu + ((a.u >> 16) & 1u);
    uint32_t rb = b.u + 0x7FFFu + ((b.u >> 16) & 1u);
    return (ra >> 16) | (rb & 0xFFFF0000u);
}

// ---------------------------------------------------------------------------
// Kernel 1: row-normalize fp32 -> bf16.  One wave32 per row, lane owns 16 elems.
// ---------------------------------------------------------------------------
__global__ __launch_bounds__(256) void k_normalize(const float* __restrict__ emb,
                                                   unsigned short* __restrict__ ubf) {
    const int wave = threadIdx.x >> 5;
    const int lane = threadIdx.x & 31;
    const int row  = blockIdx.x * 8 + wave;

    const float* rp = emb + (size_t)row * N_DIM + lane * 16;
    float v[16];
    #pragma unroll
    for (int c = 0; c < 4; ++c) {
        float4 t = ((const float4*)rp)[c];
        v[4 * c + 0] = t.x; v[4 * c + 1] = t.y;
        v[4 * c + 2] = t.z; v[4 * c + 3] = t.w;
    }
    float ss = 0.0f;
    #pragma unroll
    for (int i = 0; i < 16; ++i) ss += v[i] * v[i];

    // full-wave butterfly reduction (wave32)
    #pragma unroll
    for (int m = 1; m < 32; m <<= 1) ss += __shfl_xor(ss, m, 32);

    const float scale = 1.0f / fmaxf(sqrtf(ss), EPSV);

    unsigned int o[8];
    #pragma unroll
    for (int i = 0; i < 8; ++i)
        o[i] = pack_bf16_rne(v[2 * i] * scale, v[2 * i + 1] * scale);

    uint4* wp = (uint4*)(ubf + (size_t)row * N_DIM + lane * 16);
    wp[0] = make_uint4(o[0], o[1], o[2], o[3]);
    wp[1] = make_uint4(o[4], o[5], o[6], o[7]);
}

// ---------------------------------------------------------------------------
// Kernel 2: fused  sim = (U U^T) * INV_T  ->  exp -> row-sums + diagonal.
// Block = 256 threads (8 waves), owns 16 rows.  A fragments (16 rows x K=512)
// live in 128 VGPRs for the whole kernel.  B fragments are software-pipelined
// through a 4-deep rotating register buffer so up to 3 fragments (6 b128
// loads) are in flight behind each v_wmma, including across tile boundaries.
// ---------------------------------------------------------------------------
__global__ __launch_bounds__(256) void k_simexp(const unsigned short* __restrict__ ubf,
                                                float* __restrict__ rowsum,
                                                float* __restrict__ diagv) {
    __shared__ float red[8 * 16];

    const int wave    = threadIdx.x >> 5;
    const int lane    = threadIdx.x & 31;
    const int l15     = lane & 15;
    const int halfsel = lane >> 4;          // 0: K halves {0..7,16..23}, 1: {8..15,24..31}
    const int rowbase = blockIdx.x * 16;

    // ---- A fragments: lane holds row (rowbase + l15), two 16B chunks / k-step
    const unsigned short* abase =
        ubf + (size_t)(rowbase + l15) * N_DIM + (halfsel << 3);
    FragU afrag[16];
    #pragma unroll
    for (int k = 0; k < 16; ++k) {
        afrag[k].u4[0] = *(const uint4*)(abase + k * 32);
        afrag[k].u4[1] = *(const uint4*)(abase + k * 32 + 16);
    }

    float laneAcc[8];
    #pragma unroll
    for (int j = 0; j < 8; ++j) laneAcc[j] = 0.0f;

    // ---- B pipeline: 4 rotating fragment buffers, prefetch distance 3
    FragU bbuf[4];
    int colbase = wave << 4;
    const unsigned short* bbase =
        ubf + (size_t)(colbase + l15) * N_DIM + (halfsel << 3);
    load_frag(bbuf[0], bbase);
    load_frag(bbuf[1], bbase + 32);
    load_frag(bbuf[2], bbase + 64);

    // 512 column tiles / 8 waves = 64 per wave; per-wave tile stride = 128 cols
    for (int it = 0; it < 64; ++it) {
        // next tile's base row, wrapped so the final (dead) prefetches stay in-bounds
        const int nextrow = ((colbase + 128) & (N_ROWS - 1)) + l15;
        const unsigned short* nbase =
            ubf + (size_t)nextrow * N_DIM + (halfsel << 3);

        v8f acc = {};
        #pragma unroll
        for (int k = 0; k < 16; ++k) {
            // prefetch fragment k+3 (spills into next tile's frags 0..2 for k>=13)
            if (k < 13) load_frag(bbuf[(k + 3) & 3], bbase + (k + 3) * 32);
            else        load_frag(bbuf[(k + 3) & 3], nbase + (k - 13) * 32);
            acc = __builtin_amdgcn_wmma_f32_16x16x32_bf16(
                false, afrag[k].v, false, bbuf[k & 3].v, (short)0, acc, false, false);
        }

        float cv[8];
        #pragma unroll
        for (int j = 0; j < 8; ++j) cv[j] = acc[j] * INV_T;

        // diagonal tile: lane with n == m emits sim[i][i] (pre-exp, scaled)
        if (colbase == rowbase) {
            #pragma unroll
            for (int j = 0; j < 8; ++j) {
                const int m = j + (halfsel << 3);
                if (m == l15) diagv[rowbase + m] = cv[j];
            }
        }

        #pragma unroll
        for (int j = 0; j < 8; ++j) laneAcc[j] += __expf(cv[j]);

        colbase += 128;
        bbase = nbase;
    }

    // reduce across the 16 columns held by each half-wave
    #pragma unroll
    for (int j = 0; j < 8; ++j) {
        float s = laneAcc[j];
        s += __shfl_xor(s, 1, 32);
        s += __shfl_xor(s, 2, 32);
        s += __shfl_xor(s, 4, 32);
        s += __shfl_xor(s, 8, 32);
        laneAcc[j] = s;
    }
    // lane 0 holds rows 0..7, lane 16 holds rows 8..15 (per wave partials)
    if (l15 == 0) {
        #pragma unroll
        for (int j = 0; j < 8; ++j)
            red[wave * 16 + (halfsel << 3) + j] = laneAcc[j];
    }
    __syncthreads();
    if (threadIdx.x < 16) {
        float s = 0.0f;
        #pragma unroll
        for (int w = 0; w < 8; ++w) s += red[w * 16 + threadIdx.x];
        rowsum[rowbase + threadIdx.x] = s;
    }
}

// ---------------------------------------------------------------------------
// Kernel 3: loss = mean_i [ log(2*rowsum[2i] - exp(diag[2i])) - log(diag[2i+1]) ]
// (sim is symmetric => colsum(e) == rowsum(e))
// ---------------------------------------------------------------------------
__global__ __launch_bounds__(256) void k_finalize(const float* __restrict__ rowsum,
                                                  const float* __restrict__ diagv,
                                                  float* __restrict__ out) {
    __shared__ float red[256];
    float s = 0.0f;
    for (int i = threadIdx.x; i < N_ROWS / 2; i += 256) {
        const float se  = 2.0f * rowsum[2 * i] - __expf(diagv[2 * i]);
        const float pos = diagv[2 * i + 1];
        s += __logf(se) - __logf(pos);
    }
    red[threadIdx.x] = s;
    __syncthreads();
    for (int off = 128; off > 0; off >>= 1) {
        if (threadIdx.x < off) red[threadIdx.x] += red[threadIdx.x + off];
        __syncthreads();
    }
    if (threadIdx.x == 0) out[0] = red[0] / (float)(N_ROWS / 2);
}

// ---------------------------------------------------------------------------
extern "C" void kernel_launch(void* const* d_in, const int* in_sizes, int n_in,
                              void* d_out, int out_size, void* d_ws, size_t ws_size,
                              hipStream_t stream) {
    (void)in_sizes; (void)n_in; (void)out_size; (void)ws_size;

    const float* emb = (const float*)d_in[0];

    // workspace layout: [ U_bf16 : 8192*512*2 B ][ rowsum : 8192 f32 ][ diag : 8192 f32 ]
    unsigned short* ubf = (unsigned short*)d_ws;
    float* rowsum = (float*)((char*)d_ws + (size_t)N_ROWS * N_DIM * 2);
    float* diagv  = rowsum + N_ROWS;
    float* out    = (float*)d_out;

    k_normalize<<<N_ROWS / 8,  256, 0, stream>>>(emb, ubf);
    k_simexp   <<<N_ROWS / 16, 256, 0, stream>>>(ubf, rowsum, diagv);
    k_finalize <<<1,           256, 0, stream>>>(rowsum, diagv, out);
}
